// Transformer_78881369358928
// MI455X (gfx1250) — matrix operand order repaired and checked by reading, not compiled, not run on hardware
//
#include <hip/hip_runtime.h>
#include <hip/hip_bf16.h>
#include <math.h>

// ---------------------------------------------------------------------------
// CDNA5 (gfx1250) transformer forward pass.
//   - All GEMMs via v_wmma_f32_16x16x32_f16 (f16 in, f32 accumulate).
//   - wave32 everywhere; fragment layouts per CDNA5 ISA §7.12.2.
// ---------------------------------------------------------------------------

typedef _Float16 f16;
typedef __attribute__((ext_vector_type(16))) _Float16 v16h;
typedef __attribute__((ext_vector_type(8)))  _Float16 v8h;
typedef __attribute__((ext_vector_type(4)))  _Float16 v4h;
typedef __attribute__((ext_vector_type(8)))  float    v8f;

#define TM 64
#define TN 64
#define TK 32
#define LDH 48            // halves per LDS row: 96-byte stride, keeps 16B alignment
#define NEGV (-1e9f)

union FragH { v16h v; v8h h[2]; };

// ---------------------------------------------------------------------------
// GEMM: C[M,N] = act( A[M,K] @ W[K,N] + bias[N] (+ residual[M,N]) )
// Block tile 64x64, 8 waves; wave (wm,wn) owns a 16x32 sub-tile (two WMMAs).
// Requires M%64==0, N%64==0, K%32==0 (true for all shapes here: M=3200,
// N in {512,2048,32000}, K in {512,2048}).
// ---------------------------------------------------------------------------
__global__ __launch_bounds__(256) void gemm_bias_act_kernel(
    const float* __restrict__ A, const float* __restrict__ W,
    const float* __restrict__ bias, const float* __restrict__ residual,
    float* __restrict__ C, int M, int N, int K, int do_relu)
{
  __shared__ __align__(16) f16 As[TM * LDH];   // [row][k]
  __shared__ __align__(16) f16 Bs[TN * LDH];   // transposed: [n][k]

  const int tid  = threadIdx.x;
  const int lane = tid & 31;
  const int wave = tid >> 5;
  const int bm = blockIdx.x * TM;
  const int bn = blockIdx.y * TN;
  const int wm = (wave & 3) * 16;      // wave row group
  const int wn = (wave >> 2) * 32;     // wave col group (two 16-wide tiles)

  v8f acc0 = {0.f,0.f,0.f,0.f,0.f,0.f,0.f,0.f};
  v8f acc1 = acc0;

  // Fragment addressing (ISA §7.12.2, 16-bit A 16x32 / B 32x16):
  //  A: lanes 0-15 row m, halves 0..7 = K0..7, 8..15 = K16..23
  //     lanes 16-31 row m, halves 0..7 = K8..15, 8..15 = K24..31
  //  B: lanes 0-15 col n, halves 0..7 = K0..7, 8..15 = K16..23; lanes 16-31 col n, halves 0..7 = K8..15, 8..15 = K24..31
  const int arow  = wm + (lane & 15);
  const int aklo  = (lane < 16) ? 0 : 8;
  const int bklo  = (lane < 16) ? 0 : 8;
  const int bcol0 = wn + (lane & 15);
  const int bcol1 = bcol0 + 16;

  // Per-thread staging coordinates (float4 granularity).
  //  A tile: 64x32 floats = 512 float4 -> 2 per thread.
  //  W tile: 32x64 floats = 512 float4 -> 2 per thread.
  const int ar0 = tid >> 3;              // A rows for f = tid, tid+256
  const int ac4 = (tid & 7) << 2;        //   (row r = f>>3, col4 = (f&7)*4)
  const int bk0 = tid >> 4;              // W k-rows
  const int bn4 = (tid & 15) << 2;

  for (int k0 = 0; k0 < K; k0 += TK) {
    // ---- Global loads first (4x global_load_b128 per thread, one wait) ----
    float4 ta0 = *(const float4*)&A[(size_t)(bm + ar0)      * K + (k0 + ac4)];
    float4 ta1 = *(const float4*)&A[(size_t)(bm + ar0 + 32) * K + (k0 + ac4)];
    float4 tb0 = *(const float4*)&W[(size_t)(k0 + bk0)      * N + (bn + bn4)];
    float4 tb1 = *(const float4*)&W[(size_t)(k0 + bk0 + 16) * N + (bn + bn4)];

    // Uniform-condition prefetch of the next K tile (single scalar branch).
    if (k0 + TK < K) {
      __builtin_prefetch(&A[(size_t)(bm + ar0) * K + (k0 + TK + ac4)], 0, 1);
      __builtin_prefetch(&W[(size_t)(k0 + TK + bk0) * N + (bn + bn4)], 0, 1);
    }

    // ---- Convert + store to LDS ----
    v4h ha0 = { (f16)ta0.x, (f16)ta0.y, (f16)ta0.z, (f16)ta0.w };
    v4h ha1 = { (f16)ta1.x, (f16)ta1.y, (f16)ta1.z, (f16)ta1.w };
    *(v4h*)&As[(ar0)      * LDH + ac4] = ha0;   // 8B ds_store_b64, aligned
    *(v4h*)&As[(ar0 + 32) * LDH + ac4] = ha1;
    Bs[(bn4 + 0) * LDH + (bk0)]      = (f16)tb0.x;   // transposed scatter
    Bs[(bn4 + 1) * LDH + (bk0)]      = (f16)tb0.y;
    Bs[(bn4 + 2) * LDH + (bk0)]      = (f16)tb0.z;
    Bs[(bn4 + 3) * LDH + (bk0)]      = (f16)tb0.w;
    Bs[(bn4 + 0) * LDH + (bk0 + 16)] = (f16)tb1.x;
    Bs[(bn4 + 1) * LDH + (bk0 + 16)] = (f16)tb1.y;
    Bs[(bn4 + 2) * LDH + (bk0 + 16)] = (f16)tb1.z;
    Bs[(bn4 + 3) * LDH + (bk0 + 16)] = (f16)tb1.w;
    __syncthreads();

    // ---- Fragment loads (ds_load_b128, 16B aligned) + WMMA ----
    FragH fa, fb0, fb1;
    const f16* pa = &As[arow * LDH + aklo];
    fa.h[0] = *(const v8h*)(pa);
    fa.h[1] = *(const v8h*)(pa + 16);
    const f16* pb0 = &Bs[bcol0 * LDH + bklo];
    fb0.h[0] = *(const v8h*)(pb0);
    fb0.h[1] = *(const v8h*)(pb0 + 16);
    const f16* pb1 = &Bs[bcol1 * LDH + bklo];
    fb1.h[0] = *(const v8h*)(pb1);
    fb1.h[1] = *(const v8h*)(pb1 + 16);

    acc0 = __builtin_amdgcn_wmma_f32_16x16x32_f16(false, fa.v, false, fb0.v,
                                                  (short)0, acc0, false, false);
    acc1 = __builtin_amdgcn_wmma_f32_16x16x32_f16(false, fa.v, false, fb1.v,
                                                  (short)0, acc1, false, false);
    __syncthreads();
  }

  // C/D layout: VGPR r, lanes 0-15 -> (M=r, N=lane); lanes 16-31 -> (M=8+r, N=lane-16)
  const int rbase = bm + wm + ((lane < 16) ? 0 : 8);
  const int c0 = bn + wn + (lane & 15);
  const int c1 = c0 + 16;
  const float bia0 = bias ? bias[c0] : 0.f;
  const float bia1 = bias ? bias[c1] : 0.f;
  #pragma unroll
  for (int r = 0; r < 8; ++r) {
    int row = rbase + r;
    float v0 = acc0[r] + bia0;
    float v1 = acc1[r] + bia1;
    if (residual) {
      v0 += residual[(size_t)row * N + c0];
      v1 += residual[(size_t)row * N + c1];
    }
    if (do_relu) { v0 = fmaxf(v0, 0.f); v1 = fmaxf(v1, 0.f); }
    C[(size_t)row * N + c0] = v0;
    C[(size_t)row * N + c1] = v1;
  }
}

// ---------------------------------------------------------------------------
// Embedding gather + sinusoidal positional encoding.
// out[r, i] = E[tok[r], i] + (i even ? sin : cos)(pos * 10000^{-(i/2)/512})
// ---------------------------------------------------------------------------
__global__ void embed_pe_kernel(const float* __restrict__ E,
                                const int* __restrict__ tok,
                                float* __restrict__ out, int n, int Lp)
{
  int idx = blockIdx.x * blockDim.x + threadIdx.x;
  if (idx >= n * 512) return;
  int i = idx & 511;
  int r = idx >> 9;
  int pos = r % Lp;
  int t = tok[r];
  float ang = (float)pos * __powf(10000.0f, -(float)(i >> 1) * (1.0f / 512.0f));
  float pe = (i & 1) ? __cosf(ang) : __sinf(ang);
  out[idx] = E[(size_t)t * 512 + i] + pe;
}

// ---------------------------------------------------------------------------
// Attention core: one wave per (batch, head, query row). HD=64 -> 2 dims/lane.
// Writes output in the reference's 'm n l d -> m l (d n)' layout: col = d*NH+h.
// ---------------------------------------------------------------------------
__global__ __launch_bounds__(256) void attention_kernel(
    const float* __restrict__ Q, const float* __restrict__ Kb,
    const float* __restrict__ Vb, const int* __restrict__ mask,
    float* __restrict__ O, int Bn, int Lq, int Lk, int NHp, int HDp)
{
  int gw   = (blockIdx.x * blockDim.x + threadIdx.x) >> 5;
  int lane = threadIdx.x & 31;
  int lq = gw % Lq;
  int h  = (gw / Lq) % NHp;
  int b  = gw / (Lq * NHp);
  if (b >= Bn) return;

  const int D = NHp * HDp;  // 512
  const float* qp = Q + (size_t)(b * Lq + lq) * D + h * HDp;
  float q0 = qp[2 * lane], q1 = qp[2 * lane + 1];

  // Scores: lane holds s[j] for key k = j*32 + lane.
  float s[4] = {NEGV, NEGV, NEGV, NEGV};
  for (int k = 0; k < Lk; ++k) {
    const float* kp = Kb + (size_t)(b * Lk + k) * D + h * HDp;
    float part = q0 * kp[2 * lane] + q1 * kp[2 * lane + 1];
    #pragma unroll
    for (int o = 16; o; o >>= 1) part += __shfl_xor(part, o, 32);
    if ((k & 31) == lane) s[k >> 5] = part * 0.125f;  // 1/sqrt(64)
  }

  // Mask + softmax over Lk keys.
  float m = -INFINITY;
  #pragma unroll
  for (int j = 0; j < 4; ++j) {
    int k = j * 32 + lane;
    if (k < Lk) {
      if (mask[b * Lk + k] == 0) s[j] = NEGV;
      m = fmaxf(m, s[j]);
    }
  }
  #pragma unroll
  for (int o = 16; o; o >>= 1) m = fmaxf(m, __shfl_xor(m, o, 32));
  float p[4] = {0.f, 0.f, 0.f, 0.f};
  float sum = 0.f;
  #pragma unroll
  for (int j = 0; j < 4; ++j) {
    int k = j * 32 + lane;
    if (k < Lk) { p[j] = __expf(s[j] - m); sum += p[j]; }
  }
  #pragma unroll
  for (int o = 16; o; o >>= 1) sum += __shfl_xor(sum, o, 32);
  float inv = 1.f / sum;

  // O = P @ V ; each lane accumulates 2 head-dims.
  float a0 = 0.f, a1 = 0.f;
  #pragma unroll
  for (int j = 0; j < 4; ++j) {
    float pj = p[j] * inv;
    int kmax = Lk - j * 32; if (kmax > 32) kmax = 32;
    if (kmax > 0) {
      for (int t = 0; t < kmax; ++t) {
        float pk = __shfl(pj, t, 32);
        const float* vp = Vb + (size_t)(b * Lk + (j * 32 + t)) * D + h * HDp;
        a0 += pk * vp[2 * lane];
        a1 += pk * vp[2 * lane + 1];
      }
    }
  }
  float* op = O + (size_t)(b * Lq + lq) * D;
  op[(2 * lane)     * NHp + h] = a0;
  op[(2 * lane + 1) * NHp + h] = a1;
}

// ---------------------------------------------------------------------------
// In-place layernorm over rows of width 512; one wave per row (16 elems/lane).
// ---------------------------------------------------------------------------
__global__ __launch_bounds__(256) void layernorm512_kernel(
    float* __restrict__ X, const float* __restrict__ g,
    const float* __restrict__ bb, int rows)
{
  int gw   = (blockIdx.x * blockDim.x + threadIdx.x) >> 5;
  int lane = threadIdx.x & 31;
  if (gw >= rows) return;
  float* row = X + (size_t)gw * 512;
  float v[16];
  float sum = 0.f;
  #pragma unroll
  for (int i = 0; i < 4; ++i) {
    float4 t = *(const float4*)(row + lane * 16 + i * 4);
    v[i*4+0] = t.x; v[i*4+1] = t.y; v[i*4+2] = t.z; v[i*4+3] = t.w;
    sum += t.x + t.y + t.z + t.w;
  }
  #pragma unroll
  for (int o = 16; o; o >>= 1) sum += __shfl_xor(sum, o, 32);
  float mean = sum * (1.f / 512.f);
  float var = 0.f;
  #pragma unroll
  for (int i = 0; i < 16; ++i) { float d = v[i] - mean; var += d * d; }
  #pragma unroll
  for (int o = 16; o; o >>= 1) var += __shfl_xor(var, o, 32);
  float rstd = rsqrtf(var * (1.f / 512.f) + 1e-5f);
  #pragma unroll
  for (int i = 0; i < 16; ++i) {
    int c = lane * 16 + i;
    row[c] = g[c] * (v[i] - mean) * rstd + bb[c];
  }
}

// ---------------------------------------------------------------------------
// In-place row softmax, one 256-thread block per row (cols = 32000).
// ---------------------------------------------------------------------------
__global__ __launch_bounds__(256) void softmax_rows_kernel(
    float* __restrict__ X, int cols)
{
  __shared__ float sred[8];
  float* p = X + (size_t)blockIdx.x * cols;
  int tid = threadIdx.x, lane = tid & 31, wave = tid >> 5;

  float m = -INFINITY;
  for (int c = tid; c < cols; c += 256) m = fmaxf(m, p[c]);
  #pragma unroll
  for (int o = 16; o; o >>= 1) m = fmaxf(m, __shfl_xor(m, o, 32));
  if (lane == 0) sred[wave] = m;
  __syncthreads();
  m = fmaxf(fmaxf(fmaxf(sred[0], sred[1]), fmaxf(sred[2], sred[3])),
            fmaxf(fmaxf(sred[4], sred[5]), fmaxf(sred[6], sred[7])));
  __syncthreads();

  float s = 0.f;
  for (int c = tid; c < cols; c += 256) s += __expf(p[c] - m);
  #pragma unroll
  for (int o = 16; o; o >>= 1) s += __shfl_xor(s, o, 32);
  if (lane == 0) sred[wave] = s;
  __syncthreads();
  s = sred[0] + sred[1] + sred[2] + sred[3] + sred[4] + sred[5] + sred[6] + sred[7];
  float inv = 1.f / s;
  for (int c = tid; c < cols; c += 256) p[c] = __expf(p[c] - m) * inv;
}

// ---------------------------------------------------------------------------
// Host orchestration.
// Parameter order = jax.tree_util.tree_leaves(setup_inputs()) with sorted
// dict keys at each level:
//   0  dec_embed
//   1  dec_ff.b1   2 dec_ff.b2   3 dec_ff.ln_b  4 dec_ff.ln_g  5 dec_ff.w1  6 dec_ff.w2
//   7  dec_mha.bk  8 dec_mha.bo  9 dec_mha.bq  10 dec_mha.bv
//   11 dec_mha.ln_b 12 dec_mha.ln_g 13 dec_mha.wk 14 dec_mha.wo 15 dec_mha.wq 16 dec_mha.wv
//   17 enc_embed
//   18..23 enc_ff (b1,b2,ln_b,ln_g,w1,w2)
//   24..33 enc_mha (bk,bo,bq,bv,ln_b,ln_g,wk,wo,wq,wv)
//   34 fc_b  35 fc_w
//   36 inputs  37 outputs  38 input_mask  39 output_mask   (int32)
// ---------------------------------------------------------------------------
extern "C" void kernel_launch(void* const* d_in, const int* in_sizes, int n_in,
                              void* d_out, int out_size, void* d_ws, size_t ws_size,
                              hipStream_t stream)
{
  (void)in_sizes; (void)n_in; (void)out_size; (void)ws_size;

  constexpr int Bn = 32, Lp = 100, DIM = 512, NH = 8, HD = 64, HID = 2048, VOC = 32000;
  constexpr int M = Bn * Lp;  // 3200

  const float* dec_embed   = (const float*)d_in[0];
  const float* dec_ff_b1   = (const float*)d_in[1];
  const float* dec_ff_b2   = (const float*)d_in[2];
  const float* dec_ff_lnb  = (const float*)d_in[3];
  const float* dec_ff_lng  = (const float*)d_in[4];
  const float* dec_ff_w1   = (const float*)d_in[5];
  const float* dec_ff_w2   = (const float*)d_in[6];
  const float* dec_mha_bk  = (const float*)d_in[7];
  const float* dec_mha_bo  = (const float*)d_in[8];
  const float* dec_mha_bq  = (const float*)d_in[9];
  const float* dec_mha_bv  = (const float*)d_in[10];
  const float* dec_mha_lnb = (const float*)d_in[11];
  const float* dec_mha_lng = (const float*)d_in[12];
  const float* dec_mha_wk  = (const float*)d_in[13];
  const float* dec_mha_wo  = (const float*)d_in[14];
  const float* dec_mha_wq  = (const float*)d_in[15];
  const float* dec_mha_wv  = (const float*)d_in[16];
  const float* enc_embed   = (const float*)d_in[17];
  const float* enc_ff_b1   = (const float*)d_in[18];
  const float* enc_ff_b2   = (const float*)d_in[19];
  const float* enc_ff_lnb  = (const float*)d_in[20];
  const float* enc_ff_lng  = (const float*)d_in[21];
  const float* enc_ff_w1   = (const float*)d_in[22];
  const float* enc_ff_w2   = (const float*)d_in[23];
  const float* enc_mha_bk  = (const float*)d_in[24];
  const float* enc_mha_bo  = (const float*)d_in[25];
  const float* enc_mha_bq  = (const float*)d_in[26];
  const float* enc_mha_bv  = (const float*)d_in[27];
  const float* enc_mha_lnb = (const float*)d_in[28];
  const float* enc_mha_lng = (const float*)d_in[29];
  const float* enc_mha_wk  = (const float*)d_in[30];
  const float* enc_mha_wo  = (const float*)d_in[31];
  const float* enc_mha_wq  = (const float*)d_in[32];
  const float* enc_mha_wv  = (const float*)d_in[33];
  const float* fc_b        = (const float*)d_in[34];
  const float* fc_w        = (const float*)d_in[35];
  const int* inputs_tok    = (const int*)d_in[36];
  const int* outputs_tok   = (const int*)d_in[37];
  const int* input_mask    = (const int*)d_in[38];
  const int* output_mask   = (const int*)d_in[39];

  // Workspace layout (~79 MB of fp32).
  float* ws = (float*)d_ws;
  const size_t A512 = (size_t)M * DIM;
  float* encA = ws; ws += A512;
  float* encB = ws; ws += A512;
  float* decA = ws; ws += A512;
  float* decB = ws; ws += A512;
  float* bufQ = ws; ws += A512;
  float* bufK = ws; ws += A512;
  float* bufV = ws; ws += A512;
  float* bufO = ws; ws += A512;
  float* bufH = ws; ws += (size_t)M * HID;

  auto gemm = [&](const float* A, const float* W, const float* bias,
                  const float* res, float* C, int Mm, int Nn, int Kk, bool relu) {
    dim3 grid(Mm / TM, Nn / TN);
    gemm_bias_act_kernel<<<grid, 256, 0, stream>>>(A, W, bias, res, C, Mm, Nn, Kk, relu ? 1 : 0);
  };
  auto layernorm = [&](float* X, const float* g, const float* b) {
    layernorm512_kernel<<<(M * 32 + 255) / 256, 256, 0, stream>>>(X, g, b, M);
  };
  auto attn = [&](const float* Qb, const float* Kbf, const float* Vbf,
                  const int* mask, float* Ob) {
    int waves = Bn * NH * Lp;  // 25600 waves -> 3200 blocks of 8 waves
    attention_kernel<<<(waves * 32 + 255) / 256, 256, 0, stream>>>(
        Qb, Kbf, Vbf, mask, Ob, Bn, Lp, Lp, NH, HD);
  };
  auto mha = [&](const float* wq, const float* bq, const float* wk, const float* bk,
                 const float* wv, const float* bv, const float* wo, const float* bo,
                 const float* lng, const float* lnb,
                 const float* qin, const float* kvin, const int* mask, float* out) {
    gemm(qin,  wq, bq, nullptr, bufQ, M, DIM, DIM, false);
    gemm(kvin, wk, bk, nullptr, bufK, M, DIM, DIM, false);
    gemm(kvin, wv, bv, nullptr, bufV, M, DIM, DIM, false);
    attn(bufQ, bufK, bufV, mask, bufO);
    gemm(bufO, wo, bo, qin, out, M, DIM, DIM, false);   // +residual q
    layernorm(out, lng, lnb);
  };
  auto ff = [&](const float* w1, const float* b1, const float* w2, const float* b2,
                const float* lng, const float* lnb, const float* x, float* out) {
    gemm(x, w1, b1, nullptr, bufH, M, HID, DIM, true);  // relu
    gemm(bufH, w2, b2, x, out, M, DIM, HID, false);     // +residual x
    layernorm(out, lng, lnb);
  };

  // ---- Encoder ----
  embed_pe_kernel<<<(M * DIM + 255) / 256, 256, 0, stream>>>(enc_embed, inputs_tok, encA, M, Lp);
  float* enc = encA; float* encAlt = encB;
  for (int it = 0; it < 2; ++it) {
    mha(enc_mha_wq, enc_mha_bq, enc_mha_wk, enc_mha_bk, enc_mha_wv, enc_mha_bv,
        enc_mha_wo, enc_mha_bo, enc_mha_lng, enc_mha_lnb,
        enc, enc, input_mask, encAlt);
    { float* t = enc; enc = encAlt; encAlt = t; }
    ff(enc_ff_w1, enc_ff_b1, enc_ff_w2, enc_ff_b2, enc_ff_lng, enc_ff_lnb, enc, encAlt);
    { float* t = enc; enc = encAlt; encAlt = t; }
  }

  // ---- Decoder ----
  embed_pe_kernel<<<(M * DIM + 255) / 256, 256, 0, stream>>>(dec_embed, outputs_tok, decA, M, Lp);
  float* dec = decA; float* decAlt = decB;
  for (int it = 0; it < 2; ++it) {
    // self-attention over decoder tokens
    mha(dec_mha_wq, dec_mha_bq, dec_mha_wk, dec_mha_bk, dec_mha_wv, dec_mha_bv,
        dec_mha_wo, dec_mha_bo, dec_mha_lng, dec_mha_lnb,
        dec, dec, output_mask, decAlt);
    { float* t = dec; dec = decAlt; decAlt = t; }
    // cross-attention: queries from decoder, keys/values from encoder output
    mha(dec_mha_wq, dec_mha_bq, dec_mha_wk, dec_mha_bk, dec_mha_wv, dec_mha_bv,
        dec_mha_wo, dec_mha_bo, dec_mha_lng, dec_mha_lnb,
        dec, enc, input_mask, decAlt);
    { float* t = dec; dec = decAlt; decAlt = t; }
    ff(dec_ff_w1, dec_ff_b1, dec_ff_w2, dec_ff_b2, dec_ff_lng, dec_ff_lnb, dec, decAlt);
    { float* t = dec; dec = decAlt; decAlt = t; }
  }

  // ---- Output projection + softmax (in place in d_out) ----
  float* logits = (float*)d_out;
  gemm(dec, fc_w, fc_b, nullptr, logits, M, VOC, DIM, false);
  softmax_rows_kernel<<<M, 256, 0, stream>>>(logits, VOC);
}
